// SelfAttention_8392366096665
// MI455X (gfx1250) — matrix operand order repaired and checked by reading, not compiled, not run on hardware
//
#include <hip/hip_runtime.h>

// Problem sizes (fixed by the reference)
#define BB  32
#define SS  4096
#define DD  512
#define DAA 256
#define RR  8

typedef __attribute__((ext_vector_type(16))) __bf16 v16bf;
typedef __attribute__((ext_vector_type(8)))  __bf16 v8bf;
typedef __attribute__((ext_vector_type(4)))  __bf16 v4bf;
typedef __attribute__((ext_vector_type(8)))  float  v8f;

static __device__ __forceinline__ v16bf bf_cat(v8bf a, v8bf b) {
  return __builtin_shufflevector(a, b, 0,1,2,3,4,5,6,7,8,9,10,11,12,13,14,15);
}

// ---------------------------------------------------------------------------
// Kernel 0: convert Ws1 [DA,D] and Ws2 [R,DA] fp32 -> bf16 scratch (L2-resident)
// ---------------------------------------------------------------------------
__global__ __launch_bounds__(256)
void k_convert(const float* __restrict__ ws1, const float* __restrict__ ws2,
               __bf16* __restrict__ w1b, __bf16* __restrict__ w2b) {
  const int i = blockIdx.x * 256 + threadIdx.x;
  if (i < DAA * DD) w1b[i] = (__bf16)ws1[i];
  if (i < RR * DAA) w2b[i] = (__bf16)ws2[i];
}

// ---------------------------------------------------------------------------
// Kernel Z: zero d_out (outputs + penalty are accumulated atomically)
// ---------------------------------------------------------------------------
__global__ __launch_bounds__(256)
void k_zero(float* __restrict__ p, int n) {
  const int i = blockIdx.x * 256 + threadIdx.x;
  if (i < n) p[i] = 0.0f;
}

// ---------------------------------------------------------------------------
// Kernel A: logits[b][r][s] = Ws2 @ tanh(Ws1 @ x^T)   (fused, bf16 WMMA)
//   grid = (SS/32, BB), block = 256 (8 wave32 waves)
//   Per WG: M = 32 S-rows, N = 256 (GEMM1) then N = 16 (GEMM2, cols 0..7 = r)
// ---------------------------------------------------------------------------
#define TSROWS 32
#define XPITCH  (DD + 8)    // 520 halfs/row  (row stride 1040 B, 16-B aligned)
#define D1PITCH (DAA + 8)   // 264 halfs/row  (row stride 528 B,  16-B aligned)

__global__ __launch_bounds__(256)
void k_logits(const float*  __restrict__ x,
              const __bf16* __restrict__ w1b,
              const __bf16* __restrict__ w2b,
              float*        __restrict__ logits)
{
  __shared__ __bf16 xt[TSROWS * XPITCH];    // x tile, bf16, row-major (K contiguous)
  __shared__ __bf16 d1[TSROWS * D1PITCH];   // tanh(d1) tile, bf16

  const int b    = blockIdx.y;
  const int s0   = blockIdx.x * TSROWS;
  const int tid  = threadIdx.x;
  const int lane = tid & 31;
  const int w    = tid >> 5;

  // ---- stage x tile (f32 -> bf16) into LDS, coalesced float4 reads ----
  const float4* xg = (const float4*)(x + ((size_t)b * SS + s0) * DD);
  for (int i = tid; i < TSROWS * (DD / 4); i += 256) {
    const int row = i >> 7;               // DD/4 = 128 float4 per row
    const int c4  = i & 127;
    // prefetch the next grid-stride chunk of the streaming read
    if (i + 256 < TSROWS * (DD / 4))
      __builtin_prefetch(&xg[((i + 256) >> 7) * (DD / 4) + ((i + 256) & 127)], 0, 1);
    float4 v = xg[row * (DD / 4) + c4];
    v4bf h = { (__bf16)v.x, (__bf16)v.y, (__bf16)v.z, (__bf16)v.w };
    *(v4bf*)&xt[row * XPITCH + c4 * 4] = h;
  }
  __syncthreads();

  // ---- GEMM1: 32x256x512, v_wmma_f32_16x16x32_bf16 ----
  // wave -> (row tile of 16, column quarter of 64); 4 C-frags per wave
  const int rt    = w & 1;
  const int cq    = w >> 1;
  const int arow  = rt * 16 + (lane & 15);
  const int akoff = (lane >> 4) * 8;      // A-frag K offset per ISA layout
  const int bkoff = (lane >> 4) * 8;      // B-frag K offset per ISA layout

  const v8f z8 = {0.f,0.f,0.f,0.f,0.f,0.f,0.f,0.f};
  v8f acc[4];
  #pragma unroll
  for (int t = 0; t < 4; ++t) acc[t] = z8;

  for (int k0 = 0; k0 < DD; k0 += 32) {
    // A frag: lane m = lane&15; V0-3 <- K[ak..ak+7] pairs, V4-7 <- K[ak+16..]
    const __bf16* ap = &xt[arow * XPITCH + k0 + akoff];
    v16bf af = bf_cat(*(const v8bf*)ap, *(const v8bf*)(ap + 16));
    #pragma unroll
    for (int t = 0; t < 4; ++t) {
      // B frag: lane n needs Ws1 row n, K[bk..bk+7] and K[bk+16..bk+23] (lanes>=16: bk=8)
      const int n = cq * 64 + t * 16 + (lane & 15);
      const __bf16* bp = w1b + n * DD + k0 + bkoff;
      v16bf bfr = bf_cat(*(const v8bf*)bp, *(const v8bf*)(bp + 16));
      acc[t] = __builtin_amdgcn_wmma_f32_16x16x32_bf16(
          false, af, false, bfr, (short)0, acc[t], false, false);
    }
  }

  // ---- tanh, write d1 tile to LDS as bf16 (C layout: lane->(N), vgpr j->(M)) ----
  #pragma unroll
  for (int t = 0; t < 4; ++t) {
    const int n = cq * 64 + t * 16 + (lane & 15);
    #pragma unroll
    for (int j = 0; j < 8; ++j) {
      const int m = rt * 16 + j + ((lane < 16) ? 0 : 8);
      d1[m * D1PITCH + n] = (__bf16)tanhf(acc[t][j]);
    }
  }
  __syncthreads();

  // ---- GEMM2: 32x16x256 against Ws2 (cols 0..7 valid), waves 0..1 ----
  if (w < 2) {
    const int arow2 = w * 16 + (lane & 15);
    v8f acc2 = z8;
    for (int k0 = 0; k0 < DAA; k0 += 32) {
      const __bf16* ap = &d1[arow2 * D1PITCH + k0 + akoff];
      v16bf af = bf_cat(*(const v8bf*)ap, *(const v8bf*)(ap + 16));
      v16bf bfr;
      const int n = lane & 15;
      if (n < RR) {
        const __bf16* bp = w2b + n * DAA + k0 + bkoff;
        bfr = bf_cat(*(const v8bf*)bp, *(const v8bf*)(bp + 16));
      } else {
        v8bf zz = {};
        bfr = bf_cat(zz, zz);
      }
      acc2 = __builtin_amdgcn_wmma_f32_16x16x32_bf16(
          false, af, false, bfr, (short)0, acc2, false, false);
    }
    const int n = lane & 15;
    if (n < RR) {
      #pragma unroll
      for (int j = 0; j < 8; ++j) {
        const int m = w * 16 + j + ((lane < 16) ? 0 : 8);
        logits[((size_t)(b * RR + n)) * SS + s0 + m] = acc2[j];
      }
    }
  }
}

// ---------------------------------------------------------------------------
// Kernel B: softmax over S for each (b, r).  grid = B*R, block = 256.
// ---------------------------------------------------------------------------
__global__ __launch_bounds__(256)
void k_softmax(const float* __restrict__ logits, float* __restrict__ A) {
  const int br = blockIdx.x;
  const float* L = logits + (size_t)br * SS;
  float*       O = A      + (size_t)br * SS;
  __shared__ float red[256];
  const int t = threadIdx.x;

  float mx = -3.402823466e+38f;
  for (int s = t; s < SS; s += 256) mx = fmaxf(mx, L[s]);
  red[t] = mx; __syncthreads();
  for (int off = 128; off > 0; off >>= 1) {
    if (t < off) red[t] = fmaxf(red[t], red[t + off]);
    __syncthreads();
  }
  mx = red[0]; __syncthreads();

  float sm = 0.0f;
  for (int s = t; s < SS; s += 256) sm += __expf(L[s] - mx);
  red[t] = sm; __syncthreads();
  for (int off = 128; off > 0; off >>= 1) {
    if (t < off) red[t] += red[t + off];
    __syncthreads();
  }
  const float inv = 1.0f / red[0];

  for (int s = t; s < SS; s += 256) O[s] = __expf(L[s] - mx) * inv;
}

// ---------------------------------------------------------------------------
// Kernel C: outputs[b][r][d] += sum_s A[b][r][s] * x[b][s][d]
//   grid = (SS/256, BB); each WG streams a 256-row s-chunk of x exactly once
//   (memory-bound: 4 FLOP/byte << VALU roofline), atomic f32 accumulation.
//   A-chunk is staged into LDS with GLOBAL_LOAD_ASYNC_TO_LDS_B32 (ASYNCcnt),
//   x stream uses software prefetch (global_prefetch_b8).
// ---------------------------------------------------------------------------
#define KC 256
__global__ __launch_bounds__(256)
void k_outputs(const float* __restrict__ x, const float* __restrict__ A,
               float* __restrict__ out) {
  const int b  = blockIdx.y;
  const int s0 = blockIdx.x * KC;
  const int t  = threadIdx.x;
  __shared__ float al[RR][KC];

  // ---- async DMA: A[b][r][s0..s0+255] -> LDS (one dword per lane per row) ----
  #pragma unroll
  for (int r = 0; r < RR; ++r) {
    const float* src = &A[((size_t)(b * RR + r)) * SS + s0 + t];
    const unsigned dst = (unsigned)(unsigned long long)&al[r][t]; // addr[31:0] = LDS offset
    asm volatile("global_load_async_to_lds_b32 %0, %1, off"
                 :: "v"(dst), "v"((unsigned long long)src)
                 : "memory");
  }
  asm volatile("s_wait_asynccnt 0x0" ::: "memory");
  __syncthreads();

  const int d0 = t * 2;                   // 256 threads x 2 cols = D
  float acc[RR][2];
  #pragma unroll
  for (int r = 0; r < RR; ++r) { acc[r][0] = 0.0f; acc[r][1] = 0.0f; }

  const float2* xp = (const float2*)(x + ((size_t)b * SS + s0) * DD + d0);
  for (int i = 0; i < KC; ++i) {
    // prefetch 16 s-rows (32 KB across the WG) ahead of the HBM stream
    if (i + 16 < KC)
      __builtin_prefetch(&xp[(size_t)(i + 16) * (DD / 2)], 0, 1);
    const float2 xv = xp[(size_t)i * (DD / 2)];
    #pragma unroll
    for (int r = 0; r < RR; ++r) {
      acc[r][0] = fmaf(al[r][i], xv.x, acc[r][0]);
      acc[r][1] = fmaf(al[r][i], xv.y, acc[r][1]);
    }
  }
  #pragma unroll
  for (int r = 0; r < RR; ++r) {
    atomicAdd(&out[((size_t)(b * RR + r)) * DD + d0],     acc[r][0]);
    atomicAdd(&out[((size_t)(b * RR + r)) * DD + d0 + 1], acc[r][1]);
  }
}

// ---------------------------------------------------------------------------
// Kernel D: penalty += ||A A^T - I||_F^2 / B  (per batch Gram of 8x4096)
// ---------------------------------------------------------------------------
__global__ __launch_bounds__(256)
void k_penalty(const float* __restrict__ A, float* __restrict__ pen) {
  const int b = blockIdx.x;
  const int t = threadIdx.x;
  const int p = t & 63, chunk = t >> 6;         // 64 (r,q) pairs x 4 s-chunks
  const int r = p >> 3, q = p & 7;
  const float* Ar = A + ((size_t)(b * RR + r)) * SS;
  const float* Aq = A + ((size_t)(b * RR + q)) * SS;

  float partial = 0.0f;
  const int sBeg = chunk * (SS / 4), sEnd = sBeg + (SS / 4);
  for (int s = sBeg; s < sEnd; ++s) partial = fmaf(Ar[s], Aq[s], partial);

  __shared__ float red[256];
  red[t] = partial; __syncthreads();
  if (t < 64) {
    float g = red[t] + red[t + 64] + red[t + 128] + red[t + 192]
              - ((r == q) ? 1.0f : 0.0f);
    atomicAdd(pen, g * g * (1.0f / (float)BB));   // PENALTY = 1.0
  }
}

// ---------------------------------------------------------------------------
// Launch
// ---------------------------------------------------------------------------
extern "C" void kernel_launch(void* const* d_in, const int* in_sizes, int n_in,
                              void* d_out, int out_size, void* d_ws, size_t ws_size,
                              hipStream_t stream) {
  (void)in_sizes; (void)n_in; (void)ws_size;
  const float* x   = (const float*)d_in[0];   // [B,S,D]
  const float* Ws1 = (const float*)d_in[1];   // [DA,D]
  const float* Ws2 = (const float*)d_in[2];   // [R,DA]
  float* out = (float*)d_out;                 // [B,R,D] ++ penalty scalar

  char* ws = (char*)d_ws;
  __bf16* w1b    = (__bf16*)(ws);                        //   262144 B
  __bf16* w2b    = (__bf16*)(ws + 262144);               //     4096 B
  float*  logits = (float*) (ws + 266240);               //  4194304 B
  float*  Abuf   = (float*) (ws + 266240 + 4194304);     //  4194304 B

  k_convert<<<(DAA * DD + 255) / 256, 256, 0, stream>>>(Ws1, Ws2, w1b, w2b);
  k_zero<<<(out_size + 255) / 256, 256, 0, stream>>>(out, out_size);
  k_logits<<<dim3(SS / TSROWS, BB), 256, 0, stream>>>(x, w1b, w2b, logits);
  k_softmax<<<BB * RR, 256, 0, stream>>>(logits, Abuf);
  k_outputs<<<dim3(SS / KC, BB), 256, 0, stream>>>(x, Abuf, out);
  k_penalty<<<BB, 256, 0, stream>>>(Abuf, out + (size_t)BB * RR * DD);
}
